// cnn_Lstm_att_44203803410548
// MI455X (gfx1250) — matrix operand order repaired and checked by reading, not compiled, not run on hardware
//
#include <hip/hip_runtime.h>
#include <cstdint>
#include <cstddef>

// ---------------------------------------------------------------------------
// Types for CDNA5 WMMA (wave32): v_wmma_f32_16x16x32_bf16
// ---------------------------------------------------------------------------
typedef float  v8f   __attribute__((ext_vector_type(8)));
typedef __bf16 v16bf __attribute__((ext_vector_type(16)));
typedef __bf16 bf16x8 __attribute__((ext_vector_type(8)));

#define BATCH 8192
#define TT 50
#define BN_EPS 1e-5f

// Make a 32-bit offset opaque to LICM/CSE so loads are re-issued each
// iteration (weights stream from L2 per step instead of being register-cached
// across the 32-step loop -> spills). Offset-only opacity keeps the base
// pointer traceable to the kernarg, so clang still infers GLOBAL address
// space (global_load_b128, LOADcnt only) instead of flat.
#define OPAQUE_U32(v) asm volatile("" : "+v"(v))

#define WMMA_BF16(a, b, c) \
  __builtin_amdgcn_wmma_f32_16x16x32_bf16(false, (a), false, (b), (short)0, (c), false, false)

// A/x fragment: halves at p and p+16 (16-bit A layout K pattern)
__device__ __forceinline__ v16bf ldfragA(const __bf16* p) {
  v16bf f;
  ((bf16x8*)&f)[0] = *(const bf16x8*)p;
  ((bf16x8*)&f)[1] = *(const bf16x8*)(p + 16);
  return f;
}
// B fragment (pre-swizzled): 16 contiguous bf16 per lane
__device__ __forceinline__ v16bf ldfragB(const __bf16* p) {
  v16bf f;
  ((bf16x8*)&f)[0] = *(const bf16x8*)p;
  ((bf16x8*)&f)[1] = *(const bf16x8*)(p + 8);
  return f;
}

// workspace layout (bytes, all 256-aligned)
static constexpr size_t OFF_ATTN  = 0;            // 64 f32: folded attention bilinear params
static constexpr size_t OFF_STATS = 256;          // 96 f32: bn1 sum/sq (32) + bn2 sum/sq (64)
static constexpr size_t OFF_BN1F  = 768;          // 544 f32: folded conv2 (w2*s1) + b2'
static constexpr size_t OFF_BN2F  = 3072;         // 64 f32: s2, t2
static constexpr size_t OFF_A1    = 4096;         // B*16*4 f32 (relu(conv1), pre-BN)
static constexpr size_t OFF_A2    = 2101248;      // B*32*4 f32 (relu(conv2), pre-BN)
static constexpr size_t OFF_SWHH0 = 6295552;      // l0_whh bf16 swizzled, 2 dirs
static constexpr size_t OFF_SWHH1 = 7344128;      // l1_whh bf16 swizzled, 2 dirs
static constexpr size_t OFF_SWIH1 = 8392704;      // l1_wih bf16 swizzled, 2 dirs
static constexpr size_t OFF_LAY0  = 10489856;     // B*32*512 bf16 (bilstm layer0 out)
static constexpr size_t OFF_H1F   = 278925312;    // B*256 f32 (layer1 fwd, last step)
static constexpr size_t OFF_H1B   = 287313920;    // B*256 f32 (layer1 bwd, pos 31)

__device__ __forceinline__ float rcp_(float x) { return __builtin_amdgcn_rcpf(x); }
__device__ __forceinline__ float sigmoidf_(float x) { return rcp_(1.0f + __expf(-x)); }
__device__ __forceinline__ float tanhf_(float x) {
  float e = __expf(-2.0f * x);
  return (1.0f - e) * rcp_(1.0f + e);
}

// ---------------------------------------------------------------------------
// 1) Fold each attention (Wq,bq,Wk,bk) into 4x4 bilinear form:
//    s(xi,xj) = xi^T M xj + u.xi + v.xj + c   (global set scaled by 1/sqrt(256))
// ---------------------------------------------------------------------------
__global__ void k_attn_params(const float* __restrict__ wq_g, const float* __restrict__ bq_g,
                              const float* __restrict__ wk_g, const float* __restrict__ bk_g,
                              const float* __restrict__ wq_l, const float* __restrict__ bq_l,
                              const float* __restrict__ wk_l, const float* __restrict__ bk_l,
                              float* __restrict__ attnp) {
  int t = threadIdx.x;
  int set = (t >= 32) ? 1 : 0;
  int idx = set ? t - 32 : t;
  if (idx >= 25) return;
  const float* wq = set ? wq_l : wq_g;
  const float* bq = set ? bq_l : bq_g;
  const float* wk = set ? wk_l : wk_g;
  const float* bk = set ? bk_l : bk_g;
  float scale = set ? 1.0f : 0.0625f;  // 1/sqrt(H)=1/16 for global only
  float acc = 0.f;
  if (idx < 16) {
    int f = idx >> 2, g = idx & 3;
    for (int h = 0; h < 256; ++h) acc += wq[f * 256 + h] * wk[g * 256 + h];
  } else if (idx < 20) {
    int f = idx - 16;
    for (int h = 0; h < 256; ++h) acc += wq[f * 256 + h] * bk[h];
  } else if (idx < 24) {
    int g = idx - 20;
    for (int h = 0; h < 256; ++h) acc += wk[g * 256 + h] * bq[h];
  } else {
    for (int h = 0; h < 256; ++h) acc += bq[h] * bk[h];
  }
  attnp[set * 32 + idx] = acc * scale;
}

// ---------------------------------------------------------------------------
// 2) Pre-swizzle LSTM weights to bf16 in WMMA B-fragment order.
// ---------------------------------------------------------------------------
__global__ void k_prep(const float* __restrict__ whh0, const float* __restrict__ whh1,
                       const float* __restrict__ wih1,
                       __bf16* __restrict__ swhh0, __bf16* __restrict__ swhh1,
                       __bf16* __restrict__ swih1) {
  int idx = blockIdx.x * 256 + threadIdx.x;
  const int MAT = 64 * 8 * 32 * 16;  // 262144 = 1024x256
  if (idx < 2 * MAT) {
    int dir = idx / MAT, w = idx % MAT;
    int q = w & 15, lane = (w >> 4) & 31, kt = (w >> 9) & 7, ntile = w >> 12;
    int k = kt * 32 + ((lane < 16) ? 0 : 8) + ((q < 8) ? q : q + 8);
    int n = ntile * 16 + (lane & 15);
    swhh0[idx] = (__bf16)whh0[(dir * 1024 + n) * 256 + k];
    swhh1[idx] = (__bf16)whh1[(dir * 1024 + n) * 256 + k];
  } else {
    int j = idx - 2 * MAT;
    const int MATW = 64 * 16 * 32 * 16;  // 524288 = 1024x512
    if (j < 2 * MATW) {
      int dir = j / MATW, w = j % MATW;
      int q = w & 15, lane = (w >> 4) & 31, kt = (w >> 9) & 15, ntile = w >> 13;
      int k = kt * 32 + ((lane < 16) ? 0 : 8) + ((q < 8) ? q : q + 8);
      int n = ntile * 16 + (lane & 15);
      swih1[j] = (__bf16)wih1[(dir * 1024 + n) * 512 + k];
    }
  }
}

// ---------------------------------------------------------------------------
// 3) Attention (global transposed-weights + banded local) + conv1 + relu,
//    accumulate BN1 batch statistics. One wave per batch element.
// ---------------------------------------------------------------------------
__global__ void __launch_bounds__(128)
k_attn_conv1(const float* __restrict__ x, const float* __restrict__ attnp,
             const float* __restrict__ w1, const float* __restrict__ b1,
             float* __restrict__ a1, float* __restrict__ gstats) {
  __shared__ float sc[4][TT][TT + 2];
  __shared__ float xs[4][TT][4];
  __shared__ float gl[4][TT][4];
  __shared__ float lo[4][TT][4];
  __shared__ float ap[64];
  __shared__ float ssum[16], ssq[16];
  const int tid = threadIdx.x, lane = tid & 31, wv = tid >> 5;
  const int b = blockIdx.x * 4 + wv;
  for (int i = lane; i < 200; i += 32) xs[wv][i >> 2][i & 3] = x[(size_t)b * 200 + i];
  if (tid < 64) ap[tid] = attnp[tid];
  if (tid < 16) { ssum[tid] = 0.f; ssq[tid] = 0.f; }
  __syncthreads();
  // ---- global: per-row softmax, store full matrix (read transposed later)
  for (int rr = 0; rr < 2; ++rr) {
    int r = lane + 32 * rr;
    if (r < TT) {
      float x0 = xs[wv][r][0], x1 = xs[wv][r][1], x2 = xs[wv][r][2], x3 = xs[wv][r][3];
      float base = ap[24] + ap[16] * x0 + ap[17] * x1 + ap[18] * x2 + ap[19] * x3;
      float z0 = ap[20] + x0 * ap[0] + x1 * ap[4] + x2 * ap[8] + x3 * ap[12];
      float z1 = ap[21] + x0 * ap[1] + x1 * ap[5] + x2 * ap[9] + x3 * ap[13];
      float z2 = ap[22] + x0 * ap[2] + x1 * ap[6] + x2 * ap[10] + x3 * ap[14];
      float z3 = ap[23] + x0 * ap[3] + x1 * ap[7] + x2 * ap[11] + x3 * ap[15];
      float mx = -1e30f;
      for (int c = 0; c < TT; ++c) {
        float s = base + z0 * xs[wv][c][0] + z1 * xs[wv][c][1] + z2 * xs[wv][c][2] + z3 * xs[wv][c][3];
        sc[wv][r][c] = s;
        mx = fmaxf(mx, s);
      }
      float den = 0.f;
      for (int c = 0; c < TT; ++c) { float e = __expf(sc[wv][r][c] - mx); sc[wv][r][c] = e; den += e; }
      float inv = rcp_(den);
      for (int c = 0; c < TT; ++c) sc[wv][r][c] *= inv;
    }
  }
  __syncthreads();
  // glob[i][f] = sum_j sc[j][i] * x[j][f]   (attention_weights transposed)
  for (int rr = 0; rr < 2; ++rr) {
    int i = lane + 32 * rr;
    if (i < TT) {
      float g0 = 0, g1 = 0, g2 = 0, g3 = 0;
      for (int j = 0; j < TT; ++j) {
        float w = sc[wv][j][i];
        g0 += w * xs[wv][j][0]; g1 += w * xs[wv][j][1];
        g2 += w * xs[wv][j][2]; g3 += w * xs[wv][j][3];
      }
      gl[wv][i][0] = g0; gl[wv][i][1] = g1; gl[wv][i][2] = g2; gl[wv][i][3] = g3;
    }
  }
  // ---- local: banded |i-j|<=4, per-row softmax + weighted sum in registers
  for (int rr = 0; rr < 2; ++rr) {
    int r = lane + 32 * rr;
    if (r < TT) {
      float x0 = xs[wv][r][0], x1 = xs[wv][r][1], x2 = xs[wv][r][2], x3 = xs[wv][r][3];
      float base = ap[56] + ap[48] * x0 + ap[49] * x1 + ap[50] * x2 + ap[51] * x3;
      float z0 = ap[52] + x0 * ap[32] + x1 * ap[36] + x2 * ap[40] + x3 * ap[44];
      float z1 = ap[53] + x0 * ap[33] + x1 * ap[37] + x2 * ap[41] + x3 * ap[45];
      float z2 = ap[54] + x0 * ap[34] + x1 * ap[38] + x2 * ap[42] + x3 * ap[46];
      float z3 = ap[55] + x0 * ap[35] + x1 * ap[39] + x2 * ap[43] + x3 * ap[47];
      int jlo = (r - 4 < 0) ? 0 : r - 4;
      int jhi = (r + 4 > TT - 1) ? TT - 1 : r + 4;
      int cnt = jhi - jlo + 1;
      float sv[9];
      float mx = -1e30f;
      for (int k2 = 0; k2 < cnt; ++k2) {
        int c = jlo + k2;
        float s = base + z0 * xs[wv][c][0] + z1 * xs[wv][c][1] + z2 * xs[wv][c][2] + z3 * xs[wv][c][3];
        sv[k2] = s; mx = fmaxf(mx, s);
      }
      float den = 0, a0 = 0, a1v = 0, a2v = 0, a3 = 0;
      for (int k2 = 0; k2 < cnt; ++k2) {
        int c = jlo + k2;
        float e = __expf(sv[k2] - mx);
        den += e;
        a0 += e * xs[wv][c][0]; a1v += e * xs[wv][c][1];
        a2v += e * xs[wv][c][2]; a3 += e * xs[wv][c][3];
      }
      float inv = rcp_(den);
      lo[wv][r][0] = a0 * inv; lo[wv][r][1] = a1v * inv;
      lo[wv][r][2] = a2v * inv; lo[wv][r][3] = a3 * inv;
    }
  }
  __syncthreads();
  // ---- conv1 (pointwise over 100 concat channels) + relu + BN1 stats
  for (int t = tid; t < 256; t += 128) {
    int e = t >> 6, idx = t & 63, o = idx >> 2, l = idx & 3;
    float s = b1[o];
    for (int c = 0; c < TT; ++c) s += w1[o * 100 + c] * gl[e][c][l];
    for (int c = 0; c < TT; ++c) s += w1[o * 100 + 50 + c] * lo[e][c][l];
    float v = s > 0.f ? s : 0.f;
    a1[(((size_t)blockIdx.x * 4 + e) * 16 + o) * 4 + l] = v;
    atomicAdd(&ssum[o], v);
    atomicAdd(&ssq[o], v * v);
  }
  __syncthreads();
  if (tid < 16) {
    atomicAdd(&gstats[tid], ssum[tid]);
    atomicAdd(&gstats[16 + tid], ssq[tid]);
  }
}

// ---------------------------------------------------------------------------
// 4) Finalize BN1 and fold it into conv2:  w2'[o][c]=w2*s1[c],  b2'=b2+w2@t1
// ---------------------------------------------------------------------------
__global__ void k_bn1_fold(const float* __restrict__ gstats,
                           const float* __restrict__ bn1_g, const float* __restrict__ bn1_b,
                           const float* __restrict__ w2, const float* __restrict__ b2,
                           float* __restrict__ fold) {
  __shared__ float s1[16], t1[16];
  int t = threadIdx.x;
  if (t < 16) {
    float cnt = (float)BATCH * 4.0f;
    float m = gstats[t] / cnt;
    float v = gstats[16 + t] / cnt - m * m;
    float s = bn1_g[t] * rsqrtf(v + BN_EPS);
    s1[t] = s;
    t1[t] = bn1_b[t] - m * s;
  }
  __syncthreads();
  if (t < 512) { int o = t >> 4, c = t & 15; fold[t] = w2[o * 16 + c] * s1[c]; }
  if (t < 32) {
    float acc = b2[t];
    for (int c = 0; c < 16; ++c) acc += w2[t * 16 + c] * t1[c];
    fold[512 + t] = acc;
  }
}

// ---------------------------------------------------------------------------
// 5) conv2(bn1(a1)) + relu, accumulate BN2 stats
// ---------------------------------------------------------------------------
__global__ void __launch_bounds__(256)
k_conv2(const float* __restrict__ a1, const float* __restrict__ fold,
        float* __restrict__ a2, float* __restrict__ gstats2) {
  __shared__ float ssum[32], ssq[32];
  int t = threadIdx.x;
  if (t < 32) { ssum[t] = 0.f; ssq[t] = 0.f; }
  __syncthreads();
  int e = t >> 7, idx = t & 127, o = idx >> 2, l = idx & 3;
  size_t b = (size_t)blockIdx.x * 2 + e;
  float s = fold[512 + o];
  for (int c = 0; c < 16; ++c) s += fold[o * 16 + c] * a1[(b * 16 + c) * 4 + l];
  float v = s > 0.f ? s : 0.f;
  a2[(b * 32 + o) * 4 + l] = v;
  atomicAdd(&ssum[o], v);
  atomicAdd(&ssq[o], v * v);
  __syncthreads();
  if (t < 32) {
    atomicAdd(&gstats2[t], ssum[t]);
    atomicAdd(&gstats2[32 + t], ssq[t]);
  }
}

__global__ void k_bn2(const float* __restrict__ gstats2, const float* __restrict__ bn2_g,
                      const float* __restrict__ bn2_b, float* __restrict__ bn2f) {
  int t = threadIdx.x;
  if (t >= 32) return;
  float cnt = (float)BATCH * 4.0f;
  float m = gstats2[t] / cnt;
  float v = gstats2[32 + t] / cnt - m * m;
  float s = bn2_g[t] * rsqrtf(v + BN_EPS);
  bn2f[t] = s;
  bn2f[32 + t] = bn2_b[t] - m * s;
}

// ---------------------------------------------------------------------------
// 6) BiLSTM layer 0 (one direction per launch). Block = 16 batch rows, 8 waves.
//    Per step: gates[16x1024] = h[16x256] @ whh^T via WMMA bf16, c in regs.
//    Weight B-frags: 2-deep software pipeline, streamed from L2.
// ---------------------------------------------------------------------------
__global__ void __launch_bounds__(256)
k_lstm0(const float* __restrict__ a2, const float* __restrict__ bn2f,
        const float* __restrict__ l0_wih, const float* __restrict__ l0_b,
        const __bf16* __restrict__ whh_swz, __bf16* __restrict__ lay0, int dir) {
  __shared__ float xin[16][32][4];     // bn2-applied input (time-reversed for dir=1)
  __shared__ float wih[1024][4];
  __shared__ float bias[1024];
  __shared__ __bf16 hbuf[16][256];     // current h, WMMA-A source
  const int tid = threadIdx.x, lane = tid & 31, wv = tid >> 5;
  const int half = lane >> 4, lm = lane & 15;
  const int brow0 = blockIdx.x * 16;
  for (int i = tid; i < 2048; i += 256) {
    int m = i >> 7, s = (i >> 2) & 31, f = i & 3;
    int spos = dir ? 31 - s : s;
    xin[m][s][f] = a2[(((size_t)(brow0 + m)) * 32 + spos) * 4 + f] * bn2f[spos] + bn2f[32 + spos];
  }
  for (int i = tid; i < 4096; i += 256) (&wih[0][0])[i] = l0_wih[dir * 4096 + i];
  for (int i = tid; i < 1024; i += 256) bias[i] = l0_b[dir * 1024 + i];
  for (int i = tid; i < 4096; i += 256) (&hbuf[0][0])[i] = (__bf16)0.0f;
  __syncthreads();
  const __bf16* wsw = whh_swz + (size_t)dir * (64 * 8 * 32 * 16);
  float creg[2][8];
#pragma unroll
  for (int jj = 0; jj < 2; ++jj)
#pragma unroll
    for (int r = 0; r < 8; ++r) creg[jj][r] = 0.f;

#pragma clang loop unroll(disable)
  for (int s = 0; s < 32; ++s) {
    // A fragments of h (16x256, 8 K-tiles)
    v16bf afr[8];
#pragma unroll
    for (int kt = 0; kt < 8; ++kt) {
      int kb = kt * 32 + half * 8;
      v16bf f;
      ((bf16x8*)&f)[0] = *(const bf16x8*)&hbuf[lm][kb];
      ((bf16x8*)&f)[1] = *(const bf16x8*)&hbuf[lm][kb + 16];
      afr[kt] = f;
    }
    __syncthreads();
    v8f D[4][2];
#pragma unroll
    for (int g = 0; g < 4; ++g) {
#pragma unroll
      for (int jj = 0; jj < 2; ++jj) {
        int ntile = g * 16 + 2 * wv + jj;
        unsigned off = ((unsigned)ntile * 8 * 32 + (unsigned)lane) * 16;
        OPAQUE_U32(off);  // force per-step streaming from L2, keep global AS
        const __bf16* bp = wsw + off;
        __builtin_prefetch((const void*)(bp + 4096), 0, 1);
        v8f acc = {};
        v16bf b0 = ldfragB(bp);
        v16bf b1 = ldfragB(bp + 512);
#pragma unroll
        for (int kt = 0; kt < 6; ++kt) {
          v16bf b2 = ldfragB(bp + (kt + 2) * 512);
          acc = WMMA_BF16(afr[kt], b0, acc);
          b0 = b1;
          b1 = b2;
        }
        acc = WMMA_BF16(afr[6], b0, acc);
        acc = WMMA_BF16(afr[7], b1, acc);
        D[g][jj] = acc;
      }
    }
    int spos = dir ? 31 - s : s;
#pragma unroll
    for (int r = 0; r < 8; ++r) {
      int m = half * 8 + r;
      float x0 = xin[m][s][0], x1 = xin[m][s][1], x2 = xin[m][s][2], x3 = xin[m][s][3];
#pragma unroll
      for (int jj = 0; jj < 2; ++jj) {
        int u = wv * 32 + jj * 16 + lm;
        int ni = u, nf = 256 + u, ng = 512 + u, no = 768 + u;
        float gi = D[0][jj][r] + bias[ni] + x0 * wih[ni][0] + x1 * wih[ni][1] + x2 * wih[ni][2] + x3 * wih[ni][3];
        float gf = D[1][jj][r] + bias[nf] + x0 * wih[nf][0] + x1 * wih[nf][1] + x2 * wih[nf][2] + x3 * wih[nf][3];
        float gg = D[2][jj][r] + bias[ng] + x0 * wih[ng][0] + x1 * wih[ng][1] + x2 * wih[ng][2] + x3 * wih[ng][3];
        float go = D[3][jj][r] + bias[no] + x0 * wih[no][0] + x1 * wih[no][1] + x2 * wih[no][2] + x3 * wih[no][3];
        float iv = sigmoidf_(gi), fv = sigmoidf_(gf), gv = tanhf_(gg), ov = sigmoidf_(go);
        float cn = fv * creg[jj][r] + iv * gv;
        creg[jj][r] = cn;
        float hv = ov * tanhf_(cn);
        hbuf[m][u] = (__bf16)hv;
        lay0[(((size_t)(brow0 + m)) * 32 + spos) * 512 + (size_t)dir * 256 + u] = (__bf16)hv;
      }
    }
    __syncthreads();
  }
}

// ---------------------------------------------------------------------------
// 7) BiLSTM layer 1 forward: gates = x_s[16x512]@wih1^T + h[16x256]@whh1^T.
//    Dual-stream (x-frag + w-frag) 1-deep pipeline; only last step stored.
// ---------------------------------------------------------------------------
__global__ void __launch_bounds__(256)
k_lstm1_fwd(const __bf16* __restrict__ lay0, const __bf16* __restrict__ wih_swz,
            const __bf16* __restrict__ whh_swz, const float* __restrict__ l1_b,
            float* __restrict__ h1f) {
  __shared__ float bias[1024];
  __shared__ __bf16 hbuf[16][256];
  const int tid = threadIdx.x, lane = tid & 31, wv = tid >> 5;
  const int half = lane >> 4, lm = lane & 15;
  const int brow0 = blockIdx.x * 16;
  for (int i = tid; i < 1024; i += 256) bias[i] = l1_b[i];
  for (int i = tid; i < 4096; i += 256) (&hbuf[0][0])[i] = (__bf16)0.0f;
  __syncthreads();
  float creg[2][8];
#pragma unroll
  for (int jj = 0; jj < 2; ++jj)
#pragma unroll
    for (int r = 0; r < 8; ++r) creg[jj][r] = 0.f;
  const __bf16* xbase = lay0 + (((size_t)(brow0 + lm)) * 32) * 512 + half * 8;

#pragma clang loop unroll(disable)
  for (int s = 0; s < 32; ++s) {
    v16bf hfr[8];
#pragma unroll
    for (int kt = 0; kt < 8; ++kt) {
      int kb = kt * 32 + half * 8;
      v16bf f;
      ((bf16x8*)&f)[0] = *(const bf16x8*)&hbuf[lm][kb];
      ((bf16x8*)&f)[1] = *(const bf16x8*)&hbuf[lm][kb + 16];
      hfr[kt] = f;
    }
    __syncthreads();
    v8f D[4][2];
#pragma unroll
    for (int g = 0; g < 4; ++g) {
#pragma unroll
      for (int jj = 0; jj < 2; ++jj) {
        int ntile = g * 16 + 2 * wv + jj;
        unsigned woff = ((unsigned)ntile * 16 * 32 + (unsigned)lane) * 16;
        unsigned xoff = (unsigned)s * 512;
        OPAQUE_U32(woff);
        OPAQUE_U32(xoff);
        const __bf16* bpx = wih_swz + woff;
        const __bf16* xp = xbase + xoff;
        __builtin_prefetch((const void*)(bpx + 8192), 0, 1);
        v8f acc = {};
        v16bf a0 = ldfragA(xp);
        v16bf w0 = ldfragB(bpx);
#pragma unroll
        for (int kt = 0; kt < 15; ++kt) {
          v16bf a1 = ldfragA(xp + (kt + 1) * 32);
          v16bf w1 = ldfragB(bpx + (kt + 1) * 512);
          acc = WMMA_BF16(a0, w0, acc);
          a0 = a1;
          w0 = w1;
        }
        acc = WMMA_BF16(a0, w0, acc);
        unsigned hoff = ((unsigned)ntile * 8 * 32 + (unsigned)lane) * 16;
        OPAQUE_U32(hoff);
        const __bf16* bph = whh_swz + hoff;
        v16bf h0 = ldfragB(bph);
#pragma unroll
        for (int kt = 0; kt < 7; ++kt) {
          v16bf h1 = ldfragB(bph + (kt + 1) * 512);
          acc = WMMA_BF16(hfr[kt], h0, acc);
          h0 = h1;
        }
        acc = WMMA_BF16(hfr[7], h0, acc);
        D[g][jj] = acc;
      }
    }
#pragma unroll
    for (int r = 0; r < 8; ++r) {
      int m = half * 8 + r;
#pragma unroll
      for (int jj = 0; jj < 2; ++jj) {
        int u = wv * 32 + jj * 16 + lm;
        float gi = D[0][jj][r] + bias[u];
        float gf = D[1][jj][r] + bias[256 + u];
        float gg = D[2][jj][r] + bias[512 + u];
        float go = D[3][jj][r] + bias[768 + u];
        float iv = sigmoidf_(gi), fv = sigmoidf_(gf), gv = tanhf_(gg), ov = sigmoidf_(go);
        float cn = fv * creg[jj][r] + iv * gv;
        creg[jj][r] = cn;
        float hv = ov * tanhf_(cn);
        hbuf[m][u] = (__bf16)hv;
        if (s == 31) h1f[(size_t)(brow0 + m) * 256 + u] = hv;
      }
    }
    __syncthreads();
  }
}

// ---------------------------------------------------------------------------
// 8) BiLSTM layer 1 backward, single step (pos 31 of output = scan step 0, h=c=0)
// ---------------------------------------------------------------------------
__global__ void __launch_bounds__(256)
k_lstm1_bwd(const __bf16* __restrict__ lay0, const __bf16* __restrict__ wih_swz,
            const float* __restrict__ l1_b, float* __restrict__ h1b) {
  __shared__ float bias[1024];
  const int tid = threadIdx.x, lane = tid & 31, wv = tid >> 5;
  const int half = lane >> 4, lm = lane & 15;
  const int brow0 = blockIdx.x * 16;
  for (int i = tid; i < 1024; i += 256) bias[i] = l1_b[1024 + i];
  __syncthreads();
  const __bf16* xrow = lay0 + (((size_t)(brow0 + lm)) * 32 + 31) * 512 + half * 8;
  v8f D[4][2];
#pragma unroll
  for (int g = 0; g < 4; ++g) {
#pragma unroll
    for (int jj = 0; jj < 2; ++jj) {
      int ntile = g * 16 + 2 * wv + jj;
      v8f acc = {};
      unsigned woff = (unsigned)(64 * 16 * 32 * 16) +  // dir 1
                      ((unsigned)ntile * 16 * 32 + (unsigned)lane) * 16;
      OPAQUE_U32(woff);
      const __bf16* bpx = wih_swz + woff;
      v16bf a0 = ldfragA(xrow);
      v16bf w0 = ldfragB(bpx);
#pragma unroll
      for (int kt = 0; kt < 15; ++kt) {
        v16bf a1 = ldfragA(xrow + (kt + 1) * 32);
        v16bf w1 = ldfragB(bpx + (kt + 1) * 512);
        acc = WMMA_BF16(a0, w0, acc);
        a0 = a1;
        w0 = w1;
      }
      acc = WMMA_BF16(a0, w0, acc);
      D[g][jj] = acc;
    }
  }
#pragma unroll
  for (int r = 0; r < 8; ++r) {
    int m = half * 8 + r;
#pragma unroll
    for (int jj = 0; jj < 2; ++jj) {
      int u = wv * 32 + jj * 16 + lm;
      float gi = D[0][jj][r] + bias[u];
      float gg = D[2][jj][r] + bias[512 + u];
      float go = D[3][jj][r] + bias[768 + u];
      float cn = sigmoidf_(gi) * tanhf_(gg);  // f-gate * c(=0) drops out
      float hv = sigmoidf_(go) * tanhf_(cn);
      h1b[(size_t)(brow0 + m) * 256 + u] = hv;
    }
  }
}

// ---------------------------------------------------------------------------
// 9) FC on concat(fwd_last, bwd_pos31) + softmax; outputs (logits, probs)
// ---------------------------------------------------------------------------
__global__ void __launch_bounds__(256)
k_fc(const float* __restrict__ h1f, const float* __restrict__ h1b,
     const float* __restrict__ fc_w, const float* __restrict__ fc_b,
     float* __restrict__ out) {
  __shared__ float wf[512][10];
  __shared__ float bf[10];
  int tid = threadIdx.x;
  for (int i = tid; i < 5120; i += 256) (&wf[0][0])[i] = fc_w[i];
  if (tid < 10) bf[tid] = fc_b[tid];
  __syncthreads();
  size_t b = (size_t)blockIdx.x * 256 + tid;
  float lg[10];
#pragma unroll
  for (int n = 0; n < 10; ++n) lg[n] = bf[n];
  for (int u = 0; u < 256; ++u) {
    float hv = h1f[b * 256 + u];
#pragma unroll
    for (int n = 0; n < 10; ++n) lg[n] += hv * wf[u][n];
  }
  for (int u = 0; u < 256; ++u) {
    float hv = h1b[b * 256 + u];
#pragma unroll
    for (int n = 0; n < 10; ++n) lg[n] += hv * wf[256 + u][n];
  }
  float mx = lg[0];
#pragma unroll
  for (int n = 1; n < 10; ++n) mx = fmaxf(mx, lg[n]);
  float ex[10];
  float den = 0.f;
#pragma unroll
  for (int n = 0; n < 10; ++n) { ex[n] = __expf(lg[n] - mx); den += ex[n]; }
  float inv = rcp_(den);
#pragma unroll
  for (int n = 0; n < 10; ++n) {
    out[b * 10 + n] = lg[n];
    out[(size_t)BATCH * 10 + b * 10 + n] = ex[n] * inv;
  }
}

// ---------------------------------------------------------------------------
extern "C" void kernel_launch(void* const* d_in, const int* in_sizes, int n_in,
                              void* d_out, int out_size, void* d_ws, size_t ws_size,
                              hipStream_t stream) {
  const float* x      = (const float*)d_in[0];
  const float* wq_g   = (const float*)d_in[1];
  const float* bq_g   = (const float*)d_in[2];
  const float* wk_g   = (const float*)d_in[3];
  const float* bk_g   = (const float*)d_in[4];
  const float* wq_l   = (const float*)d_in[5];
  const float* bq_l   = (const float*)d_in[6];
  const float* wk_l   = (const float*)d_in[7];
  const float* bk_l   = (const float*)d_in[8];
  const float* w1     = (const float*)d_in[9];
  const float* b1     = (const float*)d_in[10];
  const float* bn1_g  = (const float*)d_in[11];
  const float* bn1_b  = (const float*)d_in[12];
  const float* w2     = (const float*)d_in[13];
  const float* b2     = (const float*)d_in[14];
  const float* bn2_g  = (const float*)d_in[15];
  const float* bn2_b  = (const float*)d_in[16];
  const float* l0_wih = (const float*)d_in[17];
  const float* l0_whh = (const float*)d_in[18];
  const float* l0_b   = (const float*)d_in[19];
  const float* l1_wih = (const float*)d_in[20];
  const float* l1_whh = (const float*)d_in[21];
  const float* l1_b   = (const float*)d_in[22];
  const float* fc_w   = (const float*)d_in[23];
  const float* fc_b   = (const float*)d_in[24];
  char* ws = (char*)d_ws;
  float* attnp   = (float*)(ws + OFF_ATTN);
  float* gstats  = (float*)(ws + OFF_STATS);
  float* fold    = (float*)(ws + OFF_BN1F);
  float* bn2f    = (float*)(ws + OFF_BN2F);
  float* a1      = (float*)(ws + OFF_A1);
  float* a2      = (float*)(ws + OFF_A2);
  __bf16* swhh0  = (__bf16*)(ws + OFF_SWHH0);
  __bf16* swhh1  = (__bf16*)(ws + OFF_SWHH1);
  __bf16* swih1  = (__bf16*)(ws + OFF_SWIH1);
  __bf16* lay0   = (__bf16*)(ws + OFF_LAY0);
  float* h1f     = (float*)(ws + OFF_H1F);
  float* h1b     = (float*)(ws + OFF_H1B);
  float* out     = (float*)d_out;

  hipMemsetAsync(gstats, 0, 96 * sizeof(float), stream);
  k_attn_params<<<1, 64, 0, stream>>>(wq_g, bq_g, wk_g, bk_g, wq_l, bq_l, wk_l, bk_l, attnp);
  k_prep<<<6144, 256, 0, stream>>>(l0_whh, l1_whh, l1_wih, swhh0, swhh1, swih1);
  k_attn_conv1<<<BATCH / 4, 128, 0, stream>>>(x, attnp, w1, b1, a1, gstats);
  k_bn1_fold<<<1, 512, 0, stream>>>(gstats, bn1_g, bn1_b, w2, b2, fold);
  k_conv2<<<BATCH / 2, 256, 0, stream>>>(a1, fold, a2, gstats + 32);
  k_bn2<<<1, 32, 0, stream>>>(gstats + 32, bn2_g, bn2_b, bn2f);
  k_lstm0<<<BATCH / 16, 256, 0, stream>>>(a2, bn2f, l0_wih, l0_b, swhh0, lay0, 0);
  k_lstm0<<<BATCH / 16, 256, 0, stream>>>(a2, bn2f, l0_wih, l0_b, swhh0, lay0, 1);
  k_lstm1_fwd<<<BATCH / 16, 256, 0, stream>>>(lay0, swih1, swhh1, l1_b, h1f);
  k_lstm1_bwd<<<BATCH / 16, 256, 0, stream>>>(lay0, swih1, l1_b, h1b);
  k_fc<<<BATCH / 256, 256, 0, stream>>>(h1f, h1b, fc_w, fc_b, out);
  (void)in_sizes; (void)n_in; (void)out_size; (void)ws_size;
}
